// GcnnSovnet_79267916415339
// MI455X (gfx1250) — matrix operand order repaired and checked by reading, not compiled, not run on hardware
//
#include <hip/hip_runtime.h>
#include <hip/hip_bf16.h>

// ---------------------------------------------------------------------------
// GcnnSovnet forward on MI455X (gfx1250, wave32), NHWC / channels-last.
// Convs = implicit GEMM on v_wmma_f32_16x16x32_f16 (fp16 in, fp32 acc).
// GEMM:  Y[M][Cout], M = NB*OH*OW,  K ordered (fh, fw, ci) so each 32-wide
// K-chunk is one contiguous 64B channel run (Cin % 32 == 0 for all big convs).
// Weight tiles: TDM tensor_load_to_lds DMA (TENSORcnt).  A tiles: vector
// global loads -> ds_store_b128.  Block tile 128x64, 4 waves, 8 WMMA/K-step.
// ---------------------------------------------------------------------------

typedef __attribute__((ext_vector_type(16))) _Float16 v16h;
typedef __attribute__((ext_vector_type(8)))  _Float16 v8h;
typedef __attribute__((ext_vector_type(8)))  float    v8f;
typedef unsigned int u32;
typedef __attribute__((ext_vector_type(4))) u32 v4u;
typedef __attribute__((ext_vector_type(8))) int v8i;
typedef __attribute__((ext_vector_type(4))) int v4i;

__device__ __forceinline__ float selu_f(float x) {
    const float a = 1.6732632423543772f, s = 1.0507009873554805f;
    return s * (x > 0.f ? x : a * (__expf(x) - 1.f));
}

// TDM 2D tile load: 32 fp16 (64B) x 64 rows from Wt[row][Kp] into LDS.
__device__ __forceinline__ void tdm_load_wtile(const _Float16* gptr, unsigned lds_byte,
                                               int Kp, int Cout) {
    unsigned long long ga = (unsigned long long)(uintptr_t)gptr;
    v4u g0;
    g0.x = 1u;                                          // count=1, user mode
    g0.y = lds_byte;                                    // lds_addr
    g0.z = (u32)ga;                                     // global_addr[31:0]
    g0.w = (u32)((ga >> 32) & 0x1FFFFFFu) | (2u << 30); // global_addr[56:32] | type=2
    v8i g1;
    g1[0] = (int)(1u << 16);                            // data_size = 2 bytes
    g1[1] = (int)(((u32)Kp & 0xFFFFu) << 16);           // tensor_dim0[15:0]
    g1[2] = (int)(((u32)Kp >> 16) | (((u32)Cout & 0xFFFFu) << 16));
    g1[3] = (int)(((u32)Cout >> 16) | (32u << 16));     // tile_dim0 = 32
    g1[4] = 64;                                         // tile_dim1 = 64
    g1[5] = (int)(u32)Kp;                               // tensor_dim0_stride
    g1[6] = 0;
    g1[7] = 0;
    v4i g2 = {0, 0, 0, 0};
    v4i g3 = {0, 0, 0, 0};
#if defined(__clang_major__) && __clang_major__ >= 23
    v8i gx = {0, 0, 0, 0, 0, 0, 0, 0};
    __builtin_amdgcn_tensor_load_to_lds(g0, g1, g2, g3, gx, 0);
#else
    __builtin_amdgcn_tensor_load_to_lds(g0, g1, g2, g3, 0);
#endif
}

// ---------------------------------------------------------------------------
// Weight pack: rotated filter bank, fp32 -> fp16, Wt[N][Kp], K = (fh,fw,ci)
// order, zero pad K..Kp.  N = co*4 + r.  grp==0: z2 (w: (co,Cin,ks,ks));
// grp==1: P4 (w: (co,Cig,4,ks,ks), input-rotation axis shifted by r).
// ---------------------------------------------------------------------------
__global__ void pack_weights_kernel(const float* __restrict__ w,
                                    _Float16* __restrict__ Wt,
                                    int cing, int ks, int grp,
                                    int K, int Kp, long total) {
    long idx = blockIdx.x * (long)blockDim.x + threadIdx.x;
    if (idx >= total) return;
    int n = (int)(idx / Kp);
    int k = (int)(idx - (long)n * Kp);
    if (k >= K) { Wt[idx] = (_Float16)0; return; }
    int co = n >> 2, r = n & 3;
    int Cint = grp ? cing * 4 : cing;
    int fhw = k / Cint;
    int c_in = k - fhw * Cint;
    int i = fhw / ks, j = fhw - (fhw / ks) * ks;
    // out[i][j] of rot90^r(src): iterate (i,j) <- (j, ks-1-i) r times.
    for (int t = 0; t < r; ++t) { int ni = j, nj = ks - 1 - i; i = ni; j = nj; }
    float val;
    if (grp) {
        int cig = c_in >> 2, s = c_in & 3;
        int s_src = (s - r) & 3;
        val = w[((((long)co * cing + cig) * 4 + s_src) * ks + i) * ks + j];
    } else {
        val = w[(((long)co * cing + c_in) * ks + i) * ks + j];
    }
    Wt[idx] = (_Float16)val;
}

__global__ void cvt_f32_f16_kernel(const float* __restrict__ in,
                                   _Float16* __restrict__ out, long n) {
    long i = blockIdx.x * (long)blockDim.x + threadIdx.x;
    if (i < n) out[i] = (_Float16)in[i];
}

// ---------------------------------------------------------------------------
// Implicit-GEMM conv (NHWC).  X: fp16 (NB,H,W,Cin).  Wt: fp16 [Cout][Kp].
// Y: fp32 (NB,OH,OW,Cout) == [M][Cout].  bias indexed channel>>2.
// Block 128 thr = 4 waves; tile 128(M)x64(N); wave 32x64 -> 8 WMMA/K-step.
// ---------------------------------------------------------------------------
__launch_bounds__(128)
__global__ void conv_wmma_kernel(const _Float16* __restrict__ X,
                                 const _Float16* __restrict__ Wt,
                                 const float* __restrict__ bias,
                                 float* __restrict__ Y,
                                 int NB, int Cin, int H, int W,
                                 int Cout, int K, int Kp, int ks, int pad,
                                 int stride, int OH, int OW) {
    __shared__ alignas(64) _Float16 ldsW[64 * 32];    // [n_local][k_local]
    __shared__ alignas(64) _Float16 ldsA[128 * 32];   // [m_local][k_local]

    const int M   = NB * OH * OW;
    const int OHW = OH * OW;

    const int tile_m = blockIdx.x * 128;
    const int tile_n = blockIdx.y * 64;
    const int tid  = threadIdx.x;
    const int lane = tid & 31;
    const int wid  = tid >> 5;
    const int wm   = wid * 32;             // wave M offset in block tile
    const int hi   = lane >> 4;
    const int lo   = lane & 15;

    // A staging: one thread = one M row (32 K-halves per step). Decode once.
    const int sm_row = tile_m + tid;
    const bool sval = sm_row < M;
    int sb = sm_row / OHW;
    int srm = sm_row - sb * OHW;
    const int soh = srm / OW;
    const int sow = srm - (srm / OW) * OW;
    const bool vecC = (Cin & 31) == 0;     // 32-wide K chunk = one (fh,fw) tap

    v8f acc[2][4] = {};

    for (int kk = 0; kk < K; kk += 32) {
        __syncthreads();                   // previous iteration fully consumed

        if (wid == 0) {
            // DMA 64x32 fp16 weight tile into LDS (overlaps A staging).
            tdm_load_wtile(Wt + (size_t)tile_n * Kp + kk,
                           (unsigned)(uintptr_t)&ldsW[0], Kp, Cout);
        }

        // ---- Stage this thread's A row into LDS ----
        {
            v8h t0 = {}, t1 = {}, t2 = {}, t3 = {};
            if (vecC) {
                int fhw = kk / Cin;
                int ci0 = kk - fhw * Cin;
                int fh = fhw / ks, fw = fhw - (fhw / ks) * ks;
                int ih = soh * stride - pad + fh;
                int iw = sow * stride - pad + fw;
                if (sval && (unsigned)ih < (unsigned)H && (unsigned)iw < (unsigned)W) {
                    const v8h* src =
                        (const v8h*)&X[((size_t)(sb * H + ih) * W + iw) * Cin + ci0];
                    t0 = src[0]; t1 = src[1]; t2 = src[2]; t3 = src[3];  // 4x b128
                }
            } else {
                // tiny-Cin fallback (z2 convs, K <= 9)
                int fhw = kk / Cin;
                int ci = kk - fhw * Cin;
                int fh = fhw / ks, fw = fhw - (fhw / ks) * ks;
                alignas(16) _Float16 tb[32];
#pragma unroll
                for (int e = 0; e < 32; ++e) {
                    _Float16 v = (_Float16)0;
                    if (sval && (kk + e) < K) {
                        int ih = soh * stride - pad + fh;
                        int iw = sow * stride - pad + fw;
                        if ((unsigned)ih < (unsigned)H && (unsigned)iw < (unsigned)W)
                            v = X[((size_t)(sb * H + ih) * W + iw) * Cin + ci];
                    }
                    tb[e] = v;
                    ++ci;
                    if (ci == Cin) { ci = 0; ++fw; if (fw == ks) { fw = 0; ++fh; } }
                }
                t0 = ((v8h*)tb)[0]; t1 = ((v8h*)tb)[1];
                t2 = ((v8h*)tb)[2]; t3 = ((v8h*)tb)[3];
            }
            v8h* dst = (v8h*)&ldsA[tid * 32];
            dst[0] = t0; dst[1] = t1; dst[2] = t2; dst[3] = t3;  // 4x ds_store_b128
        }

        if (wid == 0) __builtin_amdgcn_s_wait_tensorcnt(0);
        __syncthreads();

        // ---- Fragments from LDS ----
        // A 16x32: lane lo = row, lane half 'hi' selects K-octets.
        v16h afr[2];
#pragma unroll
        for (int tm = 0; tm < 2; ++tm) {
            int arow = wm + tm * 16 + lo;
            v8h p0 = *(const v8h*)&ldsA[arow * 32 + hi * 8];
            v8h p1 = *(const v8h*)&ldsA[arow * 32 + hi * 8 + 16];
            afr[tm] = __builtin_shufflevector(p0, p1, 0, 1, 2, 3, 4, 5, 6, 7,
                                              8, 9, 10, 11, 12, 13, 14, 15);
        }
        // B 32x16: lane lo = column, lane half 'hi' selects K-octets.
        v16h bfr[4];
#pragma unroll
        for (int bn = 0; bn < 4; ++bn) {
            v8h q0 = *(const v8h*)&ldsW[(bn * 16 + lo) * 32 + hi * 8];
            v8h q1 = *(const v8h*)&ldsW[(bn * 16 + lo) * 32 + hi * 8 + 16];
            bfr[bn] = __builtin_shufflevector(q0, q1, 0, 1, 2, 3, 4, 5, 6, 7,
                                              8, 9, 10, 11, 12, 13, 14, 15);
        }

#pragma unroll
        for (int tm = 0; tm < 2; ++tm)
#pragma unroll
            for (int bn = 0; bn < 4; ++bn)
                acc[tm][bn] = __builtin_amdgcn_wmma_f32_16x16x32_f16(
                    false, afr[tm], false, bfr[bn], (short)0, acc[tm][bn],
                    false, false);
    }

    // ---- Epilogue: Y[m*Cout + n], lane-contiguous stores ----
#pragma unroll
    for (int tm = 0; tm < 2; ++tm) {
#pragma unroll
        for (int bn = 0; bn < 4; ++bn) {
            int n = tile_n + bn * 16 + lo;
            float bv = bias ? bias[n >> 2] : 0.f;
#pragma unroll
            for (int r = 0; r < 8; ++r) {
                int m = tile_m + wm + tm * 16 + hi * 8 + r;
                if (m < M) Y[(size_t)m * Cout + n] = acc[tm][bn][r] + bv;
            }
        }
    }
}

// ---------------------------------------------------------------------------
// BatchNorm3d batch stats per 4-rotation channel group (NHWC: 4 consecutive
// channels).  One block per group co.
// ---------------------------------------------------------------------------
__global__ void bn_stats_kernel(const float* __restrict__ Y, float* __restrict__ stats,
                                long M, int C) {
    int co = blockIdx.x;
    float s = 0.f, ss = 0.f;
    for (long m = threadIdx.x; m < M; m += blockDim.x) {
        float4 v = *(const float4*)&Y[m * (long)C + 4 * co];
        s  += v.x + v.y + v.z + v.w;
        ss += v.x * v.x + v.y * v.y + v.z * v.z + v.w * v.w;
    }
    __shared__ float sh0[256], sh1[256];
    sh0[threadIdx.x] = s; sh1[threadIdx.x] = ss;
    __syncthreads();
    for (int off = 128; off; off >>= 1) {
        if ((int)threadIdx.x < off) {
            sh0[threadIdx.x] += sh0[threadIdx.x + off];
            sh1[threadIdx.x] += sh1[threadIdx.x + off];
        }
        __syncthreads();
    }
    if (threadIdx.x == 0) {
        float cnt  = (float)(M * 4);
        float mean = sh0[0] / cnt;
        float var  = sh1[0] / cnt - mean * mean;
        stats[2 * co]     = mean;
        stats[2 * co + 1] = rsqrtf(var + 1e-5f);
    }
}

__global__ void bn_apply_kernel(float* __restrict__ Y, const float* __restrict__ stats,
                                const float* __restrict__ g, const float* __restrict__ b,
                                _Float16* __restrict__ oh, long total, int C, int mode) {
    long i = blockIdx.x * (long)blockDim.x + threadIdx.x;
    if (i >= total) return;
    int c  = (int)(i % C);
    int co = c >> 2;
    float z = (Y[i] - stats[2 * co]) * stats[2 * co + 1] * g[co] + b[co];
    if (mode) z = selu_f(z);
    Y[i] = z;
    if (oh) oh[i] = (_Float16)z;
}

__global__ void selu_kernel(float* __restrict__ Y, _Float16* __restrict__ oh, long total) {
    long i = blockIdx.x * (long)blockDim.x + threadIdx.x;
    if (i >= total) return;
    float z = selu_f(Y[i]);
    Y[i] = z;
    if (oh) oh[i] = (_Float16)z;
}

__global__ void add_selu_kernel(const float* __restrict__ A, const float* __restrict__ B,
                                float* __restrict__ O, _Float16* __restrict__ oh, long total) {
    long i = blockIdx.x * (long)blockDim.x + threadIdx.x;
    if (i >= total) return;
    float z = selu_f(A[i] + B[i]);
    O[i] = z;
    if (oh) oh[i] = (_Float16)z;
}

// squash along middle axis of (outer, L, inner); in-place safe.
__global__ void squash_kernel(const float* __restrict__ in, float* __restrict__ out,
                              _Float16* __restrict__ oh, long outer, int L, long inner) {
    long p = blockIdx.x * (long)blockDim.x + threadIdx.x;
    if (p >= outer * inner) return;
    long o = p / inner, ii = p - o * inner;
    float ns = 0.f;
    for (int l = 0; l < L; ++l) {
        float v = in[(o * L + l) * inner + ii];
        ns += v * v;
    }
    float f = (ns / (1.f + ns)) * rsqrtf(ns + 1e-16f);
    for (int l = 0; l < L; ++l) {
        long idx = (o * L + l) * inner + ii;
        float v = in[idx] * f;
        out[idx] = v;
        if (oh) oh[idx] = (_Float16)v;
    }
}

// ---------------------------------------------------------------------------
// Primary-capsule squash + capsule/batch split (NHWC permutation, not in
// place):  in (B, HW, CAP, D, G)  ->  out (B*CAP, HW, D, G), squash over D.
// ---------------------------------------------------------------------------
__global__ void squash_split_kernel(const float* __restrict__ in, float* __restrict__ out,
                                    _Float16* __restrict__ oh,
                                    int B, long HW, int CAP, int D, int G) {
    long p = blockIdx.x * (long)blockDim.x + threadIdx.x;
    long total = (long)B * HW * CAP * G;
    if (p >= total) return;
    int g = (int)(p % G); long t = p / G;
    int cap = (int)(t % CAP); t /= CAP;
    long hw = t % HW;
    int b = (int)(t / HW);
    long ibase = (((long)b * HW + hw) * CAP + cap) * (long)(D * G) + g;
    long obase = (((long)(b * CAP + cap) * HW + hw) * D) * (long)G + g;
    float ns = 0.f;
    for (int dd = 0; dd < D; ++dd) {
        float v = in[ibase + (long)dd * G];
        ns += v * v;
    }
    float f = (ns / (1.f + ns)) * rsqrtf(ns + 1e-16f);
    for (int dd = 0; dd < D; ++dd) {
        float v = in[ibase + (long)dd * G] * f;
        long oi = obase + (long)dd * G;
        out[oi] = v;
        if (oh) oh[oi] = (_Float16)v;
    }
}

// ---------------------------------------------------------------------------
// Degree routing (cosine affinity + softmax over n_in=5), NHWC.
// U: (32*5, HW, n_out*D*G), channel = (j*D+dd)*G+g.
// S: (32*n_out, HW, D*G), channel = dd*G+g  (next conv's NHWC input order;
// for HW==1 this is exactly (b, j, dd, g) = reference output order).
// ---------------------------------------------------------------------------
#define NIN 5
__global__ void routing_kernel(const float* __restrict__ U, float* __restrict__ S,
                               int Bb, int n_out, int D, int G, long HW) {
    long p = blockIdx.x * (long)blockDim.x + threadIdx.x;
    long total = (long)Bb * n_out * G * HW;
    if (p >= total) return;
    long hw = p % HW; long t = p / HW;
    int g = (int)(t % G); t /= G;
    int j = (int)(t % n_out);
    int b = (int)(t / n_out);
    long Ct = (long)n_out * D * G;

    float dot[NIN][NIN];
#pragma unroll
    for (int a = 0; a < NIN; ++a)
#pragma unroll
        for (int c = 0; c < NIN; ++c) dot[a][c] = 0.f;

    for (int dd = 0; dd < D; ++dd) {
        float u[NIN];
#pragma unroll
        for (int i = 0; i < NIN; ++i)
            u[i] = U[((long)(b * NIN + i) * HW + hw) * Ct + ((long)(j * D + dd)) * G + g];
#pragma unroll
        for (int a = 0; a < NIN; ++a)
#pragma unroll
            for (int c = 0; c < NIN; ++c) dot[a][c] += u[a] * u[c];
    }
    float pre[NIN];
#pragma unroll
    for (int a = 0; a < NIN; ++a) {
        float ns = fmaxf(dot[a][a], 1e-8f);
        float sm = 0.f;
#pragma unroll
        for (int c = 0; c < NIN; ++c) sm += dot[a][c];
        pre[a] = sm / ns;
    }
    float mx = pre[0];
#pragma unroll
    for (int a = 1; a < NIN; ++a) mx = fmaxf(mx, pre[a]);
    float e[NIN], se = 0.f;
#pragma unroll
    for (int a = 0; a < NIN; ++a) { e[a] = __expf(pre[a] - mx); se += e[a]; }
    float inv = 1.f / se;

    for (int dd = 0; dd < D; ++dd) {
        float sacc = 0.f;
#pragma unroll
        for (int i = 0; i < NIN; ++i)
            sacc += e[i] * inv *
                    U[((long)(b * NIN + i) * HW + hw) * Ct + ((long)(j * D + dd)) * G + g];
        S[((long)(b * n_out + j) * HW + hw) * (long)(D * G) + (long)dd * G + g] = sacc;
    }
}

// ---------------------------------------------------------------------------
// Host orchestration
// ---------------------------------------------------------------------------
static inline int conv_out_dim(int H, int k, int pad, int s) { return (H + 2 * pad - k) / s + 1; }

extern "C" void kernel_launch(void* const* d_in, const int* in_sizes, int n_in,
                              void* d_out, int out_size, void* d_ws, size_t ws_size,
                              hipStream_t stream) {
    (void)in_sizes; (void)n_in; (void)out_size; (void)ws_size;
    // JAX tree-flatten order (dicts sorted by key): x, target, then params:
    //   cls{b,w}, hc1..hc5{b1,b2,bs,c1b,c1w,c2b,c2w,g1,g2,gs,scb,scw},
    //   init{same 12}, prim{b,cb,cw,g, rb{12}}
    auto P = [&](int i) { return (const float*)d_in[i]; };
    const int I_X = 0;
    const int I_CLS_B = 2, I_CLS_W = 3;
    const int I_HC = 4;
    const int I_INIT = 64;
    const int I_PRIM = 76;

    struct RB {
        const float *b1, *b2, *bs, *c1b, *c1w, *c2b, *c2w, *g1, *g2, *gs, *scb, *scw;
    };
    auto getRB = [&](int base) {
        RB r;
        r.b1 = P(base + 0);  r.b2 = P(base + 1);  r.bs = P(base + 2);
        r.c1b = P(base + 3); r.c1w = P(base + 4); r.c2b = P(base + 5);
        r.c2w = P(base + 6); r.g1 = P(base + 7);  r.g2 = P(base + 8);
        r.gs = P(base + 9);  r.scb = P(base + 10); r.scw = P(base + 11);
        return r;
    };

    // Workspace carve (max activation: 160x14x14x1280 fp32)
    char* w = (char*)d_ws;
    const size_t SZ_F32 = 40140800ull * 4;
    const size_t SZ_F16 = 32112640ull * 2;
    const size_t SZ_WT  = 5898240ull  * 2;
    float*    A     = (float*)(w);
    float*    Bf    = (float*)(w + SZ_F32);
    _Float16* XA    = (_Float16*)(w + 2 * SZ_F32);
    _Float16* XB    = (_Float16*)(w + 2 * SZ_F32 + SZ_F16);
    _Float16* Wt    = (_Float16*)(w + 2 * SZ_F32 + 2 * SZ_F16);
    float*    stats = (float*)(w + 2 * SZ_F32 + 2 * SZ_F16 + SZ_WT);

    auto pack = [&](const float* wsrc, int co, int cing, int ks, int grp) {
        int K = (grp ? cing * 4 : cing) * ks * ks;
        int Kp = (K + 31) & ~31;
        long total = (long)co * 4 * Kp;
        pack_weights_kernel<<<(unsigned)((total + 255) / 256), 256, 0, stream>>>(
            wsrc, Wt, cing, ks, grp, K, Kp, total);
        return Kp;
    };
    auto conv = [&](const _Float16* Xh, const float* bias, float* Y,
                    int NB, int Cin, int H, int Cout, int ks, int pad, int stride,
                    int Kp) {
        int OH = conv_out_dim(H, ks, pad, stride);
        int K = Cin * ks * ks;
        int M = NB * OH * OH;
        dim3 grid((M + 127) / 128, Cout / 64);
        conv_wmma_kernel<<<grid, 128, 0, stream>>>(Xh, Wt, bias, Y, NB, Cin, H, H,
                                                   Cout, K, Kp, ks, pad, stride, OH, OH);
        return OH;
    };
    auto bn = [&](float* Y, int NB, int co_g, int OH, const float* g, const float* b,
                  int mode, _Float16* oh) {
        long M = (long)NB * OH * OH;
        int C = co_g * 4;
        bn_stats_kernel<<<co_g, 256, 0, stream>>>(Y, stats, M, C);
        long total = M * C;
        bn_apply_kernel<<<(unsigned)((total + 255) / 256), 256, 0, stream>>>(
            Y, stats, g, b, oh, total, C, mode);
    };
    auto resblock = [&](const _Float16* in_h, _Float16* tmp_h, RB p,
                        int NB, int H, int stride) {
        int Kp = pack(p.c1w, 64, 64, 3, 1);
        int Ho = conv(in_h, p.c1b, A, NB, 256, H, 256, 3, 1, stride, Kp);
        bn(A, NB, 64, Ho, p.g1, p.b1, 1, tmp_h);
        Kp = pack(p.c2w, 320, 64, 3, 1);
        conv(tmp_h, p.c2b, Bf, NB, 256, Ho, 1280, 3, 1, 1, Kp);
        bn(Bf, NB, 320, Ho, p.g2, p.b2, 0, nullptr);
        Kp = pack(p.scw, 320, 64, 1, 1);
        conv(in_h, p.scb, A, NB, 256, H, 1280, 1, 0, stride, Kp);
        bn(A, NB, 320, Ho, p.gs, p.bs, 0, nullptr);
        long total = (long)NB * 1280 * Ho * Ho;
        add_selu_kernel<<<(unsigned)((total + 255) / 256), 256, 0, stream>>>(
            A, Bf, A, nullptr, total);
        return Ho;
    };

    // ---- x -> fp16 (NCHW (32,1,28,28) == NHWC since C==1) ----
    cvt_f32_f16_kernel<<<(25088 + 255) / 256, 256, 0, stream>>>(P(I_X), XA, 25088);

    // ---- initial block ----
    int Kp = pack(P(I_INIT + 4), 64, 1, 3, 0);
    conv(XA, P(I_INIT + 3), A, 32, 1, 28, 256, 3, 1, 1, Kp);
    bn(A, 32, 64, 28, P(I_INIT + 7), P(I_INIT + 0), 1, XB);
    Kp = pack(P(I_INIT + 6), 64, 64, 3, 1);
    conv(XB, P(I_INIT + 5), Bf, 32, 256, 28, 256, 3, 1, 1, Kp);
    bn(Bf, 32, 64, 28, P(I_INIT + 8), P(I_INIT + 1), 0, nullptr);
    Kp = pack(P(I_INIT + 11), 64, 1, 1, 0);
    conv(XA, P(I_INIT + 10), A, 32, 1, 28, 256, 1, 0, 1, Kp);
    bn(A, 32, 64, 28, P(I_INIT + 9), P(I_INIT + 2), 0, nullptr);
    {
        long total = 32L * 256 * 784;
        add_selu_kernel<<<(unsigned)((total + 255) / 256), 256, 0, stream>>>(
            A, Bf, A, XA, total);
    }

    // ---- primary caps: bn(selu(conv)) then res_block ----
    Kp = pack(P(I_PRIM + 2), 64, 64, 3, 1);
    conv(XA, P(I_PRIM + 1), Bf, 32, 256, 28, 256, 3, 1, 1, Kp);
    {
        long total = 32L * 256 * 784;
        selu_kernel<<<(unsigned)((total + 255) / 256), 256, 0, stream>>>(Bf, nullptr, total);
    }
    bn(Bf, 32, 64, 28, P(I_PRIM + 3), P(I_PRIM + 0), 0, XB);
    resblock(XB, XA, getRB(80), 32, 28, 1);   // -> A NHWC (32,28,28,1280)
    // squash over capsule dim + split caps into batch: (32,784,5,64,4) -> (160,784,64,4)
    {
        long total = 32L * 784 * 5 * 4;
        squash_split_kernel<<<(unsigned)((total + 255) / 256), 256, 0, stream>>>(
            A, Bf, XA, 32, 784, 5, 64, 4);
    }

    // ---- hidden routing layers ----
    const int strides[5] = {2, 1, 2, 1, 3};
    int H = 28;
    for (int i = 0; i < 5; ++i) {
        RB rp = getRB(I_HC + 12 * i);
        int Ho = resblock(XA, XB, rp, 160, H, strides[i]);   // u in A (160,Ho,Ho,1280)
        long HW = (long)Ho * Ho;
        long pts = 32L * 5 * 4 * HW;
        routing_kernel<<<(unsigned)((pts + 255) / 256), 256, 0, stream>>>(
            A, Bf, 32, 5, 64, 4, HW);
        // squash over G: Bf is (160, HW, 64, 4); outer=(160*HW*64), L=4, inner=1
        long sthreads = 160L * HW * 64;
        squash_kernel<<<(unsigned)((sthreads + 255) / 256), 256, 0, stream>>>(
            Bf, Bf, XA, sthreads, 4, 1);
        H = Ho;
    }

    // ---- class capsules: p4_conv_p4 (640,64,4,3,3), pad 0 on 3x3 -> 1x1 ----
    Kp = pack(P(I_CLS_W), 640, 64, 3, 1);
    conv(XA, P(I_CLS_B), A, 160, 256, 3, 2560, 3, 0, 1, Kp);  // (160,1,1,2560)
    {
        long pts = 32L * 10 * 4;
        routing_kernel<<<(unsigned)((pts + 255) / 256), 256, 0, stream>>>(
            A, Bf, 32, 10, 64, 4, 1);
        long op = 32L * 10 * 64;   // (b, j, dd, g) contiguous == reference order
        squash_kernel<<<(unsigned)((op + 255) / 256), 256, 0, stream>>>(
            Bf, (float*)d_out, nullptr, op, 4, 1);
    }
}
